// NCA_77283641524761
// MI455X (gfx1250) — matrix operand order repaired and checked by reading, not compiled, not run on hardware
//
#include <hip/hip_runtime.h>
#include <hip/hip_bf16.h>

typedef __attribute__((ext_vector_type(16))) _Float16 v16h;
typedef __attribute__((ext_vector_type(8)))  _Float16 v8h;
typedef __attribute__((ext_vector_type(8)))  float    v8f;
typedef __attribute__((ext_vector_type(4)))  float    v4f;

#define CCH   16      // channels
#define PCH   48      // perception channels
#define HID   128     // hidden width
#define WD    256
#define HT    256
#define NBAT  32

// LDS row strides in halfs; multiples of 8 halfs (16B) for b128 alignment,
// chosen so consecutive-lane b128 loads land on distinct banks.
#define PSTRIDE  88    // perception rows  (176B = 44 dwords; 44*l mod 64 distinct)
#define W1STRIDE 88    // w1 rows
#define W2STRIDE 136   // w2 rows          (272B = 68 dwords; 4*l mod 64 distinct)

#define PLD_HALFS  (WD * PSTRIDE)          // 22528
#define WT1_HALFS  (HID * W1STRIDE)        // 11264
#define WT2_HALFS  (CCH * W2STRIDE)        // 2176
#define SMEM_BYTES ((PLD_HALFS + WT1_HALFS + WT2_HALFS) * 2 + HID * 4)

// A-matrix fragment (16x32 f16): elements 0..7 -> K = hi8+0..7,
// elements 8..15 -> K = 16+hi8+0..7  (hi8 = 8*(lane>=16))
__device__ __forceinline__ v16h load_a_frag(const _Float16* p, int hi8) {
  v8h lo = *(const v8h*)(p + hi8);
  v8h hi = *(const v8h*)(p + 16 + hi8);
  return __builtin_shufflevector(lo, hi, 0,1,2,3,4,5,6,7,8,9,10,11,12,13,14,15);
}

// B-matrix fragment (32x16 f16): elements 0..7 -> K = hi8+0..7, 8..15 -> K = 16+hi8+0..7.
__device__ __forceinline__ v16h load_b_frag(const _Float16* p, int lane) {
  int base = (lane >> 4) * 8;
  v8h lo = *(const v8h*)(p + base);
  v8h hi = *(const v8h*)(p + base + 16);
  return __builtin_shufflevector(lo, hi, 0,1,2,3,4,5,6,7,8,9,10,11,12,13,14,15);
}

__global__ __launch_bounds__(256)
void nca_step_kernel(const float* __restrict__ grid,
                     const float* __restrict__ noise,
                     const float* __restrict__ w1,
                     const float* __restrict__ b1,
                     const float* __restrict__ w2,
                     const float* __restrict__ b2,
                     float* __restrict__ out)
{
  extern __shared__ _Float16 smem[];
  _Float16* pld = smem;                       // [WD][PSTRIDE]  perception f16
  _Float16* wt1 = pld + PLD_HALFS;            // [HID][W1STRIDE] w1 (k-contig)
  _Float16* wt2 = wt1 + WT1_HALFS;            // [CCH][W2STRIDE] w2 (k-contig)
  float*    b1l = (float*)(wt2 + WT2_HALFS);  // [HID] f32 bias1

  const int tid = threadIdx.x;
  const int bh  = blockIdx.x;                 // 0 .. 32*256-1
  const int b   = bh >> 8;
  const int y   = bh & 255;
  const size_t plane = (size_t)HT * WD;

  // ---- stage weights + bias1 into LDS ----
  {
    int n  = tid >> 1;                        // 0..127
    int k0 = (tid & 1) * 32;
    #pragma unroll
    for (int i = 0; i < 32; ++i) {
      int k = k0 + i;
      wt1[n * W1STRIDE + k] = (k < PCH) ? (_Float16)w1[n * PCH + k]
                                        : (_Float16)0.0f;
    }
    if (tid < 128) {
      int n2 = tid >> 3, ks = (tid & 7) * 16;
      #pragma unroll
      for (int i = 0; i < 16; ++i)
        wt2[n2 * W2STRIDE + ks + i] = (_Float16)w2[n2 * HID + ks + i];
      b1l[tid] = b1[tid];
    }
  }

  // ---- perception for this (b, y) row: one pixel/thread, branch-free ----
  {
    const int x  = tid;
    const int xl = (x > 0)      ? x - 1 : 0;       // clamped (value masked)
    const int xr = (x < WD - 1) ? x + 1 : WD - 1;
    const int ym = (y > 0)      ? y - 1 : 0;       // uniform clamps
    const int yp = (y < HT - 1) ? y + 1 : HT - 1;
    const float mT = (y > 0)      ? 1.0f : 0.0f;   // row masks (uniform)
    const float mB = (y < HT - 1) ? 1.0f : 0.0f;
    const float mL = (x > 0)      ? 1.0f : 0.0f;   // column masks (per-lane)
    const float mR = (x < WD - 1) ? 1.0f : 0.0f;
    const float wTL = mT * mL, wTR = mT * mR, wBL = mB * mL, wBR = mB * mR;

    const float* gb = grid + (size_t)b * CCH * plane;
    v8h hv[8] = {};                            // 64 f16 accum (48 used, rest 0)
    #pragma unroll
    for (int c = 0; c < CCH; ++c) {
      const float* r0 = gb + (size_t)c * plane + (size_t)ym * WD;
      const float* r1 = gb + (size_t)c * plane + (size_t)y  * WD;
      const float* r2 = gb + (size_t)c * plane + (size_t)yp * WD;
      float tl = r0[xl] * wTL, tc = r0[x] * mT, tr = r0[xr] * wTR;
      float ml = r1[xl] * mL,  mc = r1[x],      mr = r1[xr] * mR;
      float bl = r2[xl] * wBL, bc = r2[x] * mB, br = r2[xr] * wBR;
      float sx = (tl + tr + 2.0f * ml - 2.0f * mr + bl - br) * 0.125f;
      float sy = (tl + 2.0f * tc + tr - bl - 2.0f * bc - br) * 0.125f;
      hv[(3 * c + 0) >> 3][(3 * c + 0) & 7] = (_Float16)mc;
      hv[(3 * c + 1) >> 3][(3 * c + 1) & 7] = (_Float16)sx;
      hv[(3 * c + 2) >> 3][(3 * c + 2) & 7] = (_Float16)sy;
    }
    _Float16* prow = pld + x * PSTRIDE;
    #pragma unroll
    for (int i = 0; i < 8; ++i)
      *(v8h*)(prow + i * 8) = hv[i];           // 8x ds_store_b128
  }
  __syncthreads();

  // ---- WMMA MLP: each wave handles 16-pixel chunks ----
  const int wv   = tid >> 5;
  const int lane = tid & 31;
  const int hi8  = (lane >> 4) * 8;
  const int nl   = lane & 15;
  const float bias2 = b2[nl];

  for (int chunk = wv; chunk < 16; chunk += 8) {
    const int px0 = chunk * 16;

    // GEMM1 (transposed): h^T[tile of 16 hid][16 pix] = W1_tile x P^T
    //   A = W1 rows (lane m = hid), B = perception (lane n = pixel).
    // B fragments: K = 0..31 and 32..63 of this pixel block.
    const _Float16* prow = pld + (size_t)(px0 + nl) * PSTRIDE;
    v16h bp0 = load_b_frag(prow,      lane);
    v16h bp1 = load_b_frag(prow + 32, lane);

    // GEMM1 D layout: lane n = pixel, element j = hid (t*16 + hi8 + j).
    // That is EXACTLY GEMM2's A-fragment layout: tile 2s -> elements 0..7,
    // tile 2s+1 -> elements 8..15 of A-fragment for K-step s. No LDS stage.
    v16h a2[4];
    #pragma unroll
    for (int t = 0; t < 8; ++t) {
      // bias1 broadcast as the WMMA C operand: C[m][n] = b1[t*16+m]
      const float* bp = b1l + t * 16 + hi8;
      v4f clo = *(const v4f*)bp;
      v4f chi = *(const v4f*)(bp + 4);
      v8f acc = __builtin_shufflevector(clo, chi, 0,1,2,3,4,5,6,7);

      const _Float16* arow = wt1 + (size_t)(t * 16 + nl) * W1STRIDE;
      v16h af0 = load_a_frag(arow,      hi8);
      v16h af1 = load_a_frag(arow + 32, hi8);
      acc = __builtin_amdgcn_wmma_f32_16x16x32_f16(false, af0, false, bp0,
                                                   (short)0, acc, false, false);
      acc = __builtin_amdgcn_wmma_f32_16x16x32_f16(false, af1, false, bp1,
                                                   (short)0, acc, false, false);
      #pragma unroll
      for (int j = 0; j < 8; ++j) {
        float hvv = acc[j];
        hvv = hvv > 0.0f ? hvv : 0.0f;        // ReLU
        a2[t >> 1][(t & 1) * 8 + j] = (_Float16)hvv;
      }
    }

    // GEMM2: update[16 pix][16 ch] = h x w2^T, K=128, A from registers.
    v8f acc2 = {};
    #pragma unroll
    for (int s = 0; s < 4; ++s) {
      v16h bf = load_b_frag(wt2 + (size_t)nl * W2STRIDE + s * 32, lane);
      acc2 = __builtin_amdgcn_wmma_f32_16x16x32_f16(false, a2[s], false, bf,
                                                    (short)0, acc2, false, false);
    }

    // epilogue: bias2, stochastic mask, residual, clip — float4 vectorized.
    // Lane's 8 outputs are 8 consecutive pixels at fixed channel nl.
    const int x0 = px0 + hi8;                 // multiple of 8 -> 16B aligned
    const size_t gi0 = ((size_t)b * CCH + nl) * plane + (size_t)y * WD + x0;
    const float* nrow = noise + (size_t)b * plane + (size_t)y * WD + x0;
    #pragma unroll
    for (int h4 = 0; h4 < 2; ++h4) {
      v4f g  = *(const v4f*)(grid + gi0 + h4 * 4);
      v4f nz = *(const v4f*)(nrow + h4 * 4);
      v4f r;
      #pragma unroll
      for (int q = 0; q < 4; ++q) {
        float msk = (nz[q] < 0.5f) ? 1.0f : 0.0f;
        float v = g[q] + (acc2[h4 * 4 + q] + bias2) * msk;
        r[q] = v < -2.0f ? -2.0f : (v > 2.0f ? 2.0f : v);
      }
      *(v4f*)(out + gi0 + h4 * 4) = r;
    }
  }
}

extern "C" void kernel_launch(void* const* d_in, const int* in_sizes, int n_in,
                              void* d_out, int out_size, void* d_ws, size_t ws_size,
                              hipStream_t stream) {
  const float* grid  = (const float*)d_in[0];
  const float* noise = (const float*)d_in[1];
  const float* w1    = (const float*)d_in[2];
  const float* b1    = (const float*)d_in[3];
  const float* w2    = (const float*)d_in[4];
  const float* b2    = (const float*)d_in[5];
  float* out = (float*)d_out;
  (void)in_sizes; (void)n_in; (void)out_size; (void)d_ws; (void)ws_size;

  dim3 gridDim(NBAT * HT);   // 8192 blocks: one (batch, row) each
  dim3 blockDim(256);        // 8 wave32
  nca_step_kernel<<<gridDim, blockDim, SMEM_BYTES, stream>>>(
      grid, noise, w1, b1, w2, b2, out);
}
